// RankMixerNSTokenizer_4922032521979
// MI455X (gfx1250) — matrix operand order repaired and checked by reading, not compiled, not run on hardware
//
#include <hip/hip_runtime.h>
#include <hip/hip_bf16.h>

// ---------------- problem constants ----------------
#define B_SZ   4096
#define NS     40
#define NM     8
#define LL     8
#define NFEAT  (NS + NM * LL)     // 104
#define VP1    10001              // V+1
#define D_EMB  64
#define TOTAL  3072               // (NS+NM)*D
#define HID    768                // TOTAL/4
#define T_TOK  16
#define CHUNK  192                // TOTAL/T
#define DM     512
#define LN_EPS 1e-5f

// LDS staging: B tile = 64 rows x 32 bf16, padded to 40 elems (80B) per row
#define ROW_E  40
#define BUF_E  (64 * ROW_E)       // one buffer, elements

// ---------------- types ----------------
typedef __attribute__((ext_vector_type(16))) __bf16 v16bf;
typedef __attribute__((ext_vector_type(8)))  __bf16 bf16x8;
typedef __attribute__((ext_vector_type(8)))  float  v8f;
typedef __attribute__((ext_vector_type(4)))  int    v4i;
typedef __attribute__((address_space(1))) v4i gv4i;  // global
typedef __attribute__((address_space(3))) v4i lv4i;  // LDS

__device__ __forceinline__ __bf16 f2bf(float f) {
  unsigned u = __builtin_bit_cast(unsigned, f);
  u += 0x7fffu + ((u >> 16) & 1u);                // round-to-nearest-even
  return __builtin_bit_cast(__bf16, (unsigned short)(u >> 16));
}

// ---- CDNA5 async global->LDS copy (ASYNCcnt) with portable fallback -------
__device__ __forceinline__ void async_copy16(const __bf16* g, __bf16* l) {
#if __has_builtin(__builtin_amdgcn_global_load_async_to_lds_b128)
  __builtin_amdgcn_global_load_async_to_lds_b128((gv4i*)g, (lv4i*)l, 0, 0);
#else
  *(v4i*)l = *(const v4i*)g;
#endif
}

template <int N>
__device__ __forceinline__ void wait_async() {
#if __has_builtin(__builtin_amdgcn_s_wait_asynccnt)
  __builtin_amdgcn_s_wait_asynccnt(N);
#else
  asm volatile("s_wait_asynccnt %0" ::"i"(N) : "memory");
#endif
}

// A fragment (16x32 bf16, row-major A, lda in elements).
// Layout: lanes 0-15 hold K=[0..15] in elems0-15;
// lanes 16-31 hold K=[16..31] (matches the B LDS read).
template <int LDA>
__device__ __forceinline__ v16bf load_a_frag(const __bf16* A, int m0, int k0,
                                             int lane) {
  int row  = m0 + (lane & 15);
  int koff = (lane >> 4) * 16;
  const __bf16* base = A + (size_t)row * LDA + k0 + koff;
  bf16x8 lo = *(const bf16x8*)(base);
  bf16x8 hi = *(const bf16x8*)(base + 8);
  v16bf r;
#pragma unroll
  for (int j = 0; j < 8; ++j) { r[j] = lo[j]; r[j + 8] = hi[j]; }
  return r;
}

// Block-cooperative 128(M) x 64(N) x K tile.
// 8 waves share one async-staged, double-buffered LDS B tile per k-step;
// each wave computes a 16x64 strip with 4 accumulators.
// lds points at __shared__ __bf16[2*BUF_E].
template <int K, int LDA, int LDB>
__device__ __forceinline__ void gemm_block_tile(const __bf16* __restrict__ A,
                                                const __bf16* __restrict__ BT,
                                                __bf16* lds, int m0, int n0,
                                                int lane, int tid, v8f acc[4]) {
  constexpr int NK = K / 32;
  // cooperative B staging: 256 threads x 16B = 4KB tile (64 rows x 64B)
  const int row = tid >> 2;        // 0..63  (row n0+row of BT)
  const int q   = tid & 3;         // 16-byte quarter within the 64B row chunk
  const __bf16* gsrc = BT + (size_t)(n0 + row) * LDB + q * 8;
  __bf16* ldst = lds + row * ROW_E + q * 8;

  async_copy16(gsrc, ldst);                         // stage 0 -> buffer 0
  v16bf a = load_a_frag<LDA>(A, m0, 0, lane);

#pragma unroll 2
  for (int j = 0; j < NK; ++j) {
    if (j + 1 < NK) {
      async_copy16(gsrc + (size_t)(j + 1) * 32, ldst + ((j + 1) & 1) * BUF_E);
      wait_async<1>();                              // current buffer landed
    } else {
      wait_async<0>();
    }
    __syncthreads();                                // all waves' copies visible

    v16bf an;
    if (j + 1 < NK) an = load_a_frag<LDA>(A, m0, (j + 1) * 32, lane);

    const __bf16* bb = lds + (j & 1) * BUF_E + ((lane >> 4) << 4);
    const int rsel = lane & 15;
    v16bf b0 = *(const v16bf*)(bb + (size_t)( 0 + rsel) * ROW_E);
    v16bf b1 = *(const v16bf*)(bb + (size_t)(16 + rsel) * ROW_E);
    v16bf b2 = *(const v16bf*)(bb + (size_t)(32 + rsel) * ROW_E);
    v16bf b3 = *(const v16bf*)(bb + (size_t)(48 + rsel) * ROW_E);

    acc[0] = __builtin_amdgcn_wmma_f32_16x16x32_bf16(false, a, false, b0,
                                                     (short)0, acc[0], false, false);
    acc[1] = __builtin_amdgcn_wmma_f32_16x16x32_bf16(false, a, false, b1,
                                                     (short)0, acc[1], false, false);
    acc[2] = __builtin_amdgcn_wmma_f32_16x16x32_bf16(false, a, false, b2,
                                                     (short)0, acc[2], false, false);
    acc[3] = __builtin_amdgcn_wmma_f32_16x16x32_bf16(false, a, false, b3,
                                                     (short)0, acc[3], false, false);

    __syncthreads();                                // reads done before rewrite
    if (j + 1 < NK) a = an;
  }
}

// ---------------- weight prep ----------------
// W: K x N row-major fp32  ->  WT: N x K row-major bf16
__global__ void k_transpose_bf16(const float* __restrict__ W,
                                 __bf16* __restrict__ WT, int K, int N) {
  int tid = blockIdx.x * blockDim.x + threadIdx.x;
  if (tid >= K * N) return;
  int k = tid / N, n = tid % N;
  WT[(size_t)n * K + k] = f2bf(W[tid]);
}

// proj_w: (T, CHUNK, DM) fp32 -> per-token N-major bf16 (T, DM, CHUNK)
__global__ void k_transpose_proj(const float* __restrict__ W,
                                 __bf16* __restrict__ WT) {
  int tid = blockIdx.x * blockDim.x + threadIdx.x;
  if (tid >= T_TOK * CHUNK * DM) return;
  int t = tid / (CHUNK * DM);
  int r = tid % (CHUNK * DM);
  int k = r / DM, n = r % DM;
  WT[(size_t)t * DM * CHUNK + (size_t)n * CHUNK + k] = f2bf(W[tid]);
}

// ---------------- embedding gather + mean-pool -> cat (bf16) ----------------
__global__ void k_gather_cat(const int* __restrict__ feats,
                             const float* __restrict__ es_tab,
                             const float* __restrict__ em_tab,
                             __bf16* __restrict__ cat) {
  int tid = blockIdx.x * blockDim.x + threadIdx.x;
  if (tid >= B_SZ * TOTAL) return;
  int b = tid / TOTAL;
  int c = tid % TOTAL;
  int f = c >> 6, d = c & 63;
  float v;
  if (f < NS) {
    int idx = feats[b * NFEAT + f];
    v = idx ? es_tab[((size_t)f * VP1 + idx) * D_EMB + d] : 0.0f;
  } else {
    int m = f - NS;
    float sum = 0.0f, cnt = 0.0f;
#pragma unroll
    for (int l = 0; l < LL; ++l) {
      int idx = feats[b * NFEAT + NS + m * LL + l];
      if (idx) { sum += em_tab[((size_t)m * VP1 + idx) * D_EMB + d]; cnt += 1.0f; }
    }
    v = sum / fmaxf(cnt, 1.0f);
  }
  cat[(size_t)b * TOTAL + c] = f2bf(v);
}

// ---------------- GEMM1: h = silu(cat @ w1 + b1)  (4096x3072 * 3072x768) ----
__global__ __launch_bounds__(256) void k_gemm1_silu(
    const __bf16* __restrict__ cat, const __bf16* __restrict__ w1t,
    const float* __restrict__ b1, __bf16* __restrict__ h) {
  __shared__ __align__(16) __bf16 lds_b[2 * BUF_E];
  const int tid  = threadIdx.x;
  const int lane = tid & 31;
  const int wid  = tid >> 5;
  const int NGRP = HID / 64;  // 12
  const int m0 = (blockIdx.x / NGRP) * 128 + wid * 16;
  const int n0 = (blockIdx.x % NGRP) * 64;
  v8f acc[4] = {};
  gemm_block_tile<TOTAL, TOTAL, TOTAL>(cat, w1t, lds_b, m0, n0, lane, tid, acc);
  const int mbase = m0 + ((lane >> 4) << 3);
  const int ncol  = lane & 15;
#pragma unroll
  for (int i = 0; i < 4; ++i) {
    int n = n0 + i * 16 + ncol;
    float bias = b1[n];
#pragma unroll
    for (int r = 0; r < 8; ++r) {
      float v = acc[i][r] + bias;
      float s = v / (1.0f + __expf(-v));  // silu
      h[(size_t)(mbase + r) * HID + n] = f2bf(s);
    }
  }
}

// ---- GEMM2: x = cat * sigmoid(h @ w2 + b2) * 2   (4096x768 * 768x3072) -----
__global__ __launch_bounds__(256) void k_gemm2_gate(
    const __bf16* __restrict__ h, const __bf16* __restrict__ w2t,
    const float* __restrict__ b2, const __bf16* __restrict__ cat,
    __bf16* __restrict__ x) {
  __shared__ __align__(16) __bf16 lds_b[2 * BUF_E];
  const int tid  = threadIdx.x;
  const int lane = tid & 31;
  const int wid  = tid >> 5;
  const int NGRP = TOTAL / 64;  // 48
  const int m0 = (blockIdx.x / NGRP) * 128 + wid * 16;
  const int n0 = (blockIdx.x % NGRP) * 64;
  v8f acc[4] = {};
  gemm_block_tile<HID, HID, HID>(h, w2t, lds_b, m0, n0, lane, tid, acc);
  const int mbase = m0 + ((lane >> 4) << 3);
  const int ncol  = lane & 15;
#pragma unroll
  for (int i = 0; i < 4; ++i) {
    int n = n0 + i * 16 + ncol;
    float bias = b2[n];
#pragma unroll
    for (int r = 0; r < 8; ++r) {
      int m = mbase + r;
      float g = 1.0f / (1.0f + __expf(-(acc[i][r] + bias)));
      float c = (float)cat[(size_t)m * TOTAL + n];
      x[(size_t)m * TOTAL + n] = f2bf(c * g * 2.0f);
    }
  }
}

// ---- GEMM3: y[b,t,:] = x[b,t*192:+192] @ proj_w[t] + proj_b[t]  (fp32 out) --
__global__ __launch_bounds__(256) void k_gemm3_proj(
    const __bf16* __restrict__ x, const __bf16* __restrict__ pwt,
    const float* __restrict__ proj_b, float* __restrict__ y) {
  __shared__ __align__(16) __bf16 lds_b[2 * BUF_E];
  const int tid  = threadIdx.x;
  const int lane = tid & 31;
  const int wid  = tid >> 5;
  // blocks: t (4b) x mblk (5b: 4096/128=32) x ng (3b: 512/64=8)
  const int t    = blockIdx.x >> 8;
  const int r_   = blockIdx.x & 255;
  const int m0   = (r_ >> 3) * 128 + wid * 16;
  const int n0   = (r_ & 7) * 64;
  const __bf16* A  = x + (size_t)t * CHUNK;               // row stride TOTAL
  const __bf16* BT = pwt + (size_t)t * DM * CHUNK;        // DM x CHUNK
  v8f acc[4] = {};
  gemm_block_tile<CHUNK, TOTAL, CHUNK>(A, BT, lds_b, m0, n0, lane, tid, acc);
  const int mbase = m0 + ((lane >> 4) << 3);
  const int ncol  = lane & 15;
#pragma unroll
  for (int i = 0; i < 4; ++i) {
    int n = n0 + i * 16 + ncol;
    float bias = proj_b[t * DM + n];
#pragma unroll
    for (int r = 0; r < 8; ++r) {
      int m = mbase + r;
      y[((size_t)m * T_TOK + t) * DM + n] = acc[i][r] + bias;
    }
  }
}

// ---------------- LayerNorm over DM=512, one wave per (b,t) row -------------
__global__ __launch_bounds__(256) void k_layernorm(
    float* __restrict__ y, const float* __restrict__ ln_g,
    const float* __restrict__ ln_b) {
  const int lane = threadIdx.x & 31;
  const int row  = blockIdx.x * (blockDim.x >> 5) + (threadIdx.x >> 5);
  const int t = row & (T_TOK - 1);
  float* p = y + (size_t)row * DM;
  float vals[16];
  float s = 0.0f, s2 = 0.0f;
#pragma unroll
  for (int j = 0; j < 16; ++j) {
    float v = p[lane + 32 * j];
    vals[j] = v; s += v; s2 += v * v;
  }
#pragma unroll
  for (int off = 16; off > 0; off >>= 1) {
    s  += __shfl_xor(s,  off, 32);
    s2 += __shfl_xor(s2, off, 32);
  }
  float mu  = s * (1.0f / DM);
  float var = s2 * (1.0f / DM) - mu * mu;
  float inv = rsqrtf(var + LN_EPS);
#pragma unroll
  for (int j = 0; j < 16; ++j) {
    int n = lane + 32 * j;
    p[n] = (vals[j] - mu) * inv * ln_g[t * DM + n] + ln_b[t * DM + n];
  }
}

// ---------------- host launch ----------------
extern "C" void kernel_launch(void* const* d_in, const int* in_sizes, int n_in,
                              void* d_out, int out_size, void* d_ws,
                              size_t ws_size, hipStream_t stream) {
  const int*   int_feats  = (const int*)  d_in[0];
  const float* emb_single = (const float*)d_in[1];
  const float* emb_multi  = (const float*)d_in[2];
  const float* lhuc_w1    = (const float*)d_in[3];
  const float* lhuc_b1    = (const float*)d_in[4];
  const float* lhuc_w2    = (const float*)d_in[5];
  const float* lhuc_b2    = (const float*)d_in[6];
  const float* proj_w     = (const float*)d_in[7];
  const float* proj_b     = (const float*)d_in[8];
  const float* ln_g       = (const float*)d_in[9];
  const float* ln_b       = (const float*)d_in[10];
  float* out = (float*)d_out;

  // workspace layout (bf16 buffers)
  char* ws = (char*)d_ws;
  size_t off = 0;
  __bf16* cat = (__bf16*)(ws + off); off += (size_t)B_SZ * TOTAL * 2;       // 25.2MB
  __bf16* hb  = (__bf16*)(ws + off); off += (size_t)B_SZ * HID * 2;        //  6.3MB
  __bf16* xb  = (__bf16*)(ws + off); off += (size_t)B_SZ * TOTAL * 2;      // 25.2MB
  __bf16* w1t = (__bf16*)(ws + off); off += (size_t)TOTAL * HID * 2;       //  4.7MB
  __bf16* w2t = (__bf16*)(ws + off); off += (size_t)HID * TOTAL * 2;       //  4.7MB
  __bf16* pwt = (__bf16*)(ws + off); off += (size_t)T_TOK * DM * CHUNK * 2;//  3.1MB
  (void)ws_size; (void)in_sizes; (void)n_in; (void)out_size;

  const int TB = 256;

  // weight prep
  k_transpose_bf16<<<(TOTAL * HID + TB - 1) / TB, TB, 0, stream>>>(
      lhuc_w1, w1t, TOTAL, HID);
  k_transpose_bf16<<<(HID * TOTAL + TB - 1) / TB, TB, 0, stream>>>(
      lhuc_w2, w2t, HID, TOTAL);
  k_transpose_proj<<<(T_TOK * CHUNK * DM + TB - 1) / TB, TB, 0, stream>>>(
      proj_w, pwt);

  // embedding gather + pool
  k_gather_cat<<<(B_SZ * TOTAL + TB - 1) / TB, TB, 0, stream>>>(
      int_feats, emb_single, emb_multi, cat);

  // GEMM1: (4096/128) * (768/64) = 384 blocks (block tile 128x64)
  k_gemm1_silu<<<384, TB, 0, stream>>>(cat, w1t, lhuc_b1, hb);

  // GEMM2: (4096/128) * (3072/64) = 1536 blocks
  k_gemm2_gate<<<1536, TB, 0, stream>>>(hb, w2t, lhuc_b2, cat, xb);

  // GEMM3: 16 tokens * (4096/128) * (512/64) = 4096 blocks
  k_gemm3_proj<<<4096, TB, 0, stream>>>(xb, pwt, proj_b, out);

  // LayerNorm: 4096*16 rows, one wave each
  k_layernorm<<<(B_SZ * T_TOK) / 8, TB, 0, stream>>>(out, ln_g, ln_b);
}
